// decoder1_85744727097507
// MI455X (gfx1250) — matrix operand order repaired and checked by reading, not compiled, not run on hardware
//
#include <hip/hip_runtime.h>
#include <hip/hip_bf16.h>

#define N_BATCH 128
#define T_STEPS 200
#define TX 256
#define EDIM 256
#define NMELS 80
#define OUTD 400

typedef __bf16 bf16;
typedef __attribute__((ext_vector_type(16))) bf16  v16bf;
typedef __attribute__((ext_vector_type(8)))  float v8f;

// ---------------------------------------------------------------------------
// WMMA fragment helpers (wave32; A/B use this code's contiguous k order, same for both)
// A 16x32 bf16: lane<16 -> M=lane, kb=0; lane>=16 -> M=lane-16, kb=16;
//               elements {kb..kb+15} contiguous  (2x ds_load_b128).
// B from W[out][in] row-major == column-major B: lane&15 selects N (row of W),
// lane half selects K slice 0..15 / 16..31 (2x global_load_b128).
// C/D 16x16 f32: VGPR r holds M = r + 8*(lane>=16), N = lane&15.
// ---------------------------------------------------------------------------
__device__ __forceinline__ v16bf frag_a_bf16(const bf16* A, int lda) {
  int lane = (int)(threadIdx.x & 31);
  int m    = lane & 15;
  int kb   = (lane >> 4) << 4;         // 0 or 16
  const bf16* r0 = A + m * lda + kb;
  v16bf a;
#pragma unroll
  for (int i = 0; i < 8; ++i) a[i]     = r0[i];
#pragma unroll
  for (int i = 0; i < 8; ++i) a[8 + i] = r0[8 + i];
  return a;
}

__device__ __forceinline__ v16bf frag_b_bf16(const bf16* W, int ldb) {
  int lane = (int)(threadIdx.x & 31);
  int n    = lane & 15;
  int kb   = (lane >> 4) << 4;         // 0 or 16
  const bf16* r0 = W + n * ldb + kb;
  v16bf b;
#pragma unroll
  for (int i = 0; i < 16; ++i) b[i] = r0[i];
  return b;
}

// OUT[16 x Ncols] = A[16 x K](bf16, LDS) * W[Ncols x K](bf16, global)^T + bias.
// K % 32 == 0, Ncols % 16 == 0. Distributed over nWaves wave32s; no barriers.
// B fragments are double-buffered in registers to keep one global tile load
// in flight behind each WMMA (A is on DScnt, B on LOADcnt).
template <int RELU, typename OutT>
__device__ void wg_gemm(const bf16* A, int lda, const bf16* W,
                        const float* bias, OutT* OUT, int ldo,
                        int Ncols, int K, int nWaves) {
  int wave = (int)(threadIdx.x >> 5);
  int lane = (int)(threadIdx.x & 31);
  int n    = lane & 15;
  int mo   = (lane >> 4) << 3;
  for (int nt = wave; nt < (Ncols >> 4); nt += nWaves) {
    const bf16* Wt = W + (size_t)nt * 16 * K;
    __builtin_prefetch(W + (size_t)(nt + nWaves) * 16 * K, 0, 0); // global_prefetch_b8
    v8f acc = {};
    v16bf bcur = frag_b_bf16(Wt, K);
    for (int k = 0; k < K; k += 32) {
      v16bf bnxt = bcur;
      if (k + 32 < K) bnxt = frag_b_bf16(Wt + k + 32, K);
      v16bf a = frag_a_bf16(A + k, lda);
      acc = __builtin_amdgcn_wmma_f32_16x16x32_bf16(false, a, false, bcur,
                                                    (short)0, acc, false, false);
      bcur = bnxt;
    }
    float bv = bias ? bias[nt * 16 + n] : 0.0f;
#pragma unroll
    for (int r = 0; r < 8; ++r) {
      float x = acc[r] + bv;
      if (RELU) x = fmaxf(x, 0.0f);
      OUT[(size_t)(mo + r) * ldo + nt * 16 + n] = (OutT)x;
    }
  }
}

// ---------------------------------------------------------------------------
// Precompute kernels
// ---------------------------------------------------------------------------
__global__ void cvt_f32_bf16(const float* __restrict__ src, bf16* __restrict__ dst,
                             int n) {
  int i = blockIdx.x * blockDim.x + threadIdx.x;
  if (i < n) dst[i] = (bf16)src[i];
}

// W1 [256][80] -> bf16 padded to [256][96]
__global__ void cvt_pad_w1(const float* __restrict__ src, bf16* __restrict__ dst) {
  int i = blockIdx.x * blockDim.x + threadIdx.x;
  if (i < 256 * 96) {
    int n = i / 96, k = i % 96;
    dst[i] = (bf16)((k < NMELS) ? src[n * NMELS + k] : 0.0f);
  }
}

// Fused prenet: frames(teacher-forced shift) -> relu(W1)->relu(W2) -> pre[N*T,128]
// Stage-1 output written directly as bf16 so stage-2 A-fragments need no cvt.
__global__ void prenet_kernel(const float* __restrict__ inputs,
                              const bf16* __restrict__ W1b, const float* __restrict__ b1,
                              const bf16* __restrict__ W2b, const float* __restrict__ b2,
                              float* __restrict__ pre) {
  __shared__ bf16 sAb[16][96];
  __shared__ bf16 sHb[16][256];
  int r0 = blockIdx.x * 16;
  for (int i = threadIdx.x; i < 16 * 96; i += blockDim.x) {
    int m = i / 96, k = i % 96;
    int r = r0 + m;
    int n = r / T_STEPS, t = r % T_STEPS;
    float val = 0.0f;
    if (t > 0 && k < NMELS)
      val = inputs[((size_t)n * T_STEPS + (t - 1)) * (NMELS * 5) + 4 * NMELS + k];
    sAb[m][k] = (bf16)val;
  }
  __syncthreads();
  wg_gemm<1, bf16>(&sAb[0][0], 96, W1b, b1, &sHb[0][0], 256, 256, 96, 8);
  __syncthreads();
  wg_gemm<1, float>(&sHb[0][0], 256, W2b, b2, pre + (size_t)r0 * 128, 128, 128, 256, 8);
}

// processed_memory[N*TX,256] = memory @ Wm^T
__global__ void pm_kernel(const float* __restrict__ memory,
                          const bf16* __restrict__ Wmb, float* __restrict__ pm) {
  __shared__ bf16 sAb[16][256];
  int r0 = blockIdx.x * 16;
  for (int i = threadIdx.x; i < 16 * 256; i += blockDim.x)
    sAb[i / 256][i % 256] = (bf16)memory[(size_t)r0 * 256 + i];
  __syncthreads();
  wg_gemm<0, float>(&sAb[0][0], 256, Wmb, nullptr, pm + (size_t)r0 * 256, 256, 256, 256, 8);
}

// ---------------------------------------------------------------------------
// Persistent decoder: 8 workgroups x 512 threads (16 wave32s), 16 batch rows
// each, all 200 steps inside the kernel. A operands staged once per GEMM as
// bf16 in LDS (folded into the elementwise passes). Dynamic LDS ~233 KB.
// ---------------------------------------------------------------------------
__global__ void decoder_kernel(
    const float* __restrict__ pre,     // [N*T,128]
    const float* __restrict__ pm,      // [N*TX,256]
    const float* __restrict__ memory,  // [N*TX,256]
    const float* __restrict__ vvec,    // [256]
    const bf16* Wiha, const float* bih_a, const bf16* Whha, const float* bhh_a,
    const bf16* Wqb,
    const bf16* Wpb, const float* bp,
    const bf16* Wih1, const float* bih1, const bf16* Whh1, const float* bhh1,
    const bf16* Wih2, const float* bih2, const bf16* Whh2, const float* bhh2,
    const bf16* Wob, const float* bo,
    float* __restrict__ outs, float* __restrict__ aligns) {
  extern __shared__ float sm[];
  float* gi  = sm;                // 16*768
  float* gh  = gi + 16 * 768;     // 16*768
  float* hat = gh + 16 * 768;     // 16*256
  float* ctx = hat + 16 * 256;    // 16*256
  float* h1  = ctx + 16 * 256;    // 16*256
  float* h2  = h1 + 16 * 256;     // 16*256
  float* q   = h2 + 16 * 256;     // 16*256
  float* al  = q + 16 * 256;      // 16*256
  float* dcb = al + 16 * 256;     // 16*256
  float* svv = dcb + 16 * 256;    // 256
  bf16*  xb  = (bf16*)(svv + 256);       // 16*512 bf16 A staging
  bf16*  hb  = xb + 16 * 512;            // 16*256 bf16 hidden staging

  const int NT  = (int)blockDim.x;   // 512
  const int tid = (int)threadIdx.x;
  const int n0  = (int)blockIdx.x * 16;

  for (int i = tid; i < 16 * 256; i += NT) {
    hat[i] = 0.f; ctx[i] = 0.f; h1[i] = 0.f; h2[i] = 0.f;
  }
  for (int i = tid; i < 256; i += NT) svv[i] = vvec[i];
  __syncthreads();

  for (int t = 0; t < T_STEPS; ++t) {
    // 1) stage cell_in = [prenet(x_t) | ctx] and h_att as bf16
    for (int i = tid; i < 16 * 384; i += NT) {
      int m = i / 384, c = i % 384;
      float x = (c < 128) ? pre[((size_t)(n0 + m) * T_STEPS + t) * 128 + c]
                          : ctx[m * 256 + (c - 128)];
      xb[m * 512 + c] = (bf16)x;
    }
    for (int i = tid; i < 16 * 256; i += NT) hb[i] = (bf16)hat[i];
    __syncthreads();
    // 2) attention GRU gates
    wg_gemm<0, float>(xb, 512, Wiha, bih_a, gi, 768, 768, 384, 16);
    wg_gemm<0, float>(hb, 256, Whha, bhh_a, gh, 768, 768, 256, 16);
    __syncthreads();
    // 3) h_att update (+ restage bf16 for the q GEMM)
    for (int i = tid; i < 16 * 256; i += NT) {
      int m = i >> 8, d = i & 255;
      float r  = 1.f / (1.f + __expf(-(gi[m * 768 + d]       + gh[m * 768 + d])));
      float z  = 1.f / (1.f + __expf(-(gi[m * 768 + 256 + d] + gh[m * 768 + 256 + d])));
      float nn = tanhf(gi[m * 768 + 512 + d] + r * gh[m * 768 + 512 + d]);
      float h  = (1.f - z) * nn + z * hat[i];
      hat[i] = h;
      hb[i]  = (bf16)h;
    }
    __syncthreads();
    // 4) q = h_att @ Wq^T
    wg_gemm<0, float>(hb, 256, Wqb, nullptr, q, 256, 256, 256, 16);
    __syncthreads();
    // 5) scores[m][tx] = sum_d tanh(q + pm) * v
    for (int p = tid; p < 16 * TX; p += NT) {
      int m = p >> 8, tx = p & 255;
      const float* pmr = pm + ((size_t)(n0 + m) * TX + tx) * EDIM;
      const float* qr  = q + m * 256;
      float s = 0.f;
      for (int d = 0; d < EDIM; ++d) s += tanhf(qr[d] + pmr[d]) * svv[d];
      al[m * 256 + tx] = s;
    }
    __syncthreads();
    // 6) softmax per row (one wave32 per row) + emit aligns
    {
      int w = tid >> 5, lane = tid & 31;
      float* row = al + w * 256;
      float vv[8], mx = -3.0e38f;
#pragma unroll
      for (int j = 0; j < 8; ++j) { vv[j] = row[lane + 32 * j]; mx = fmaxf(mx, vv[j]); }
      for (int o = 16; o > 0; o >>= 1) mx = fmaxf(mx, __shfl_xor(mx, o, 32));
      float s = 0.f;
#pragma unroll
      for (int j = 0; j < 8; ++j) { vv[j] = __expf(vv[j] - mx); s += vv[j]; }
      for (int o = 16; o > 0; o >>= 1) s += __shfl_xor(s, o, 32);
      float inv = 1.f / s;
      float* arow = aligns + ((size_t)(n0 + w) * T_STEPS + t) * TX;
#pragma unroll
      for (int j = 0; j < 8; ++j) {
        float a = vv[j] * inv;
        row[lane + 32 * j]  = a;
        arow[lane + 32 * j] = a;
      }
    }
    __syncthreads();
    // 7) ctx = align @ memory (per-batch mat-vec, coalesced over d)
    for (int p = tid; p < 16 * EDIM; p += NT) {
      int m = p >> 8, d = p & 255;
      const float* mr = memory + (size_t)(n0 + m) * TX * EDIM + d;
      const float* ar = al + m * 256;
      float s = 0.f;
      for (int tx = 0; tx < TX; ++tx) s += ar[tx] * mr[(size_t)tx * EDIM];
      ctx[p] = s;
    }
    __syncthreads();
    // 8) dec = [h_att | ctx] @ Wp^T + bp
    for (int i = tid; i < 16 * 512; i += NT) {
      int m = i / 512, c = i % 512;
      float x = (c < 256) ? hat[m * 256 + c] : ctx[m * 256 + (c - 256)];
      xb[i] = (bf16)x;
    }
    __syncthreads();
    wg_gemm<0, float>(xb, 512, Wpb, bp, dcb, 256, 256, 512, 16);
    __syncthreads();
    // stage dec and h1 as bf16 for GRU1 GEMMs
    for (int i = tid; i < 16 * 256; i += NT) {
      int m = i >> 8, d = i & 255;
      xb[m * 512 + d] = (bf16)dcb[i];
      hb[i] = (bf16)h1[i];
    }
    __syncthreads();
    // 9) GRU1 + residual
    wg_gemm<0, float>(xb, 512, Wih1, bih1, gi, 768, 768, 256, 16);
    wg_gemm<0, float>(hb, 256, Whh1, bhh1, gh, 768, 768, 256, 16);
    __syncthreads();
    for (int i = tid; i < 16 * 256; i += NT) {
      int m = i >> 8, d = i & 255;
      float r  = 1.f / (1.f + __expf(-(gi[m * 768 + d]       + gh[m * 768 + d])));
      float z  = 1.f / (1.f + __expf(-(gi[m * 768 + 256 + d] + gh[m * 768 + 256 + d])));
      float nn = tanhf(gi[m * 768 + 512 + d] + r * gh[m * 768 + 512 + d]);
      float h  = (1.f - z) * nn + z * h1[i];
      h1[i] = h;
      float dn = dcb[i] + h;
      dcb[i] = dn;
      xb[m * 512 + d] = (bf16)dn;     // A for GRU2 input GEMM
      hb[i] = (bf16)h2[i];            // A for GRU2 hidden GEMM
    }
    __syncthreads();
    // 10) GRU2 + residual
    wg_gemm<0, float>(xb, 512, Wih2, bih2, gi, 768, 768, 256, 16);
    wg_gemm<0, float>(hb, 256, Whh2, bhh2, gh, 768, 768, 256, 16);
    __syncthreads();
    for (int i = tid; i < 16 * 256; i += NT) {
      int m = i >> 8, d = i & 255;
      float r  = 1.f / (1.f + __expf(-(gi[m * 768 + d]       + gh[m * 768 + d])));
      float z  = 1.f / (1.f + __expf(-(gi[m * 768 + 256 + d] + gh[m * 768 + 256 + d])));
      float nn = tanhf(gi[m * 768 + 512 + d] + r * gh[m * 768 + 512 + d]);
      float h  = (1.f - z) * nn + z * h2[i];
      h2[i] = h;
      float dn = dcb[i] + h;
      dcb[i] = dn;
      xb[m * 512 + d] = (bf16)dn;     // A for output GEMM
    }
    __syncthreads();
    // 11) out = dec @ Wo^T + bo -> global (row stride T*400)
    wg_gemm<0, float>(xb, 512, Wob, bo, outs + ((size_t)n0 * T_STEPS + t) * OUTD,
                      T_STEPS * OUTD, OUTD, 256, 16);
    __syncthreads();
  }
}

// ---------------------------------------------------------------------------
extern "C" void kernel_launch(void* const* d_in, const int* in_sizes, int n_in,
                              void* d_out, int out_size, void* d_ws, size_t ws_size,
                              hipStream_t stream) {
  (void)in_sizes; (void)n_in; (void)out_size; (void)ws_size;
  const float* inputs = (const float*)d_in[0];
  const float* memory = (const float*)d_in[1];
  const float* W1 = (const float*)d_in[2];
  const float* b1 = (const float*)d_in[3];
  const float* W2 = (const float*)d_in[4];
  const float* b2 = (const float*)d_in[5];
  const float* Wm = (const float*)d_in[6];
  const float* Wq = (const float*)d_in[7];
  const float* v  = (const float*)d_in[8];
  const float* Wih_a = (const float*)d_in[9];
  const float* Whh_a = (const float*)d_in[10];
  const float* bih_a = (const float*)d_in[11];
  const float* bhh_a = (const float*)d_in[12];
  const float* Wp   = (const float*)d_in[13];
  const float* bp   = (const float*)d_in[14];
  const float* Wih1 = (const float*)d_in[15];
  const float* Whh1 = (const float*)d_in[16];
  const float* bih1 = (const float*)d_in[17];
  const float* bhh1 = (const float*)d_in[18];
  const float* Wih2 = (const float*)d_in[19];
  const float* Whh2 = (const float*)d_in[20];
  const float* bih2 = (const float*)d_in[21];
  const float* bhh2 = (const float*)d_in[22];
  const float* Wo = (const float*)d_in[23];
  const float* bo = (const float*)d_in[24];

  char* ws = (char*)d_ws;
  size_t off = 0;
  auto carve = [&](size_t bytes) {
    void* p = ws + off;
    off = (off + bytes + 255) & ~(size_t)255;
    return p;
  };

  bf16* W1b   = (bf16*)carve((size_t)256 * 96 * 2);   // padded K 80->96
  bf16* W2b   = (bf16*)carve((size_t)128 * 256 * 2);
  bf16* Wmb   = (bf16*)carve((size_t)256 * 256 * 2);
  bf16* Wqb   = (bf16*)carve((size_t)256 * 256 * 2);
  bf16* Wihab = (bf16*)carve((size_t)768 * 384 * 2);
  bf16* Whhab = (bf16*)carve((size_t)768 * 256 * 2);
  bf16* Wpb   = (bf16*)carve((size_t)256 * 512 * 2);
  bf16* Wih1b = (bf16*)carve((size_t)768 * 256 * 2);
  bf16* Whh1b = (bf16*)carve((size_t)768 * 256 * 2);
  bf16* Wih2b = (bf16*)carve((size_t)768 * 256 * 2);
  bf16* Whh2b = (bf16*)carve((size_t)768 * 256 * 2);
  bf16* Wob   = (bf16*)carve((size_t)OUTD * 256 * 2);
  float* pre  = (float*)carve((size_t)N_BATCH * T_STEPS * 128 * 4);
  float* pmb  = (float*)carve((size_t)N_BATCH * TX * EDIM * 4);

  auto cvt = [&](const float* s, bf16* d, int n) {
    cvt_f32_bf16<<<(n + 255) / 256, 256, 0, stream>>>(s, d, n);
  };
  cvt_pad_w1<<<(256 * 96 + 255) / 256, 256, 0, stream>>>(W1, W1b);
  cvt(W2, W2b, 128 * 256);
  cvt(Wm, Wmb, 256 * 256);
  cvt(Wq, Wqb, 256 * 256);
  cvt(Wih_a, Wihab, 768 * 384);
  cvt(Whh_a, Whhab, 768 * 256);
  cvt(Wp, Wpb, 256 * 512);
  cvt(Wih1, Wih1b, 768 * 256);
  cvt(Whh1, Whh1b, 768 * 256);
  cvt(Wih2, Wih2b, 768 * 256);
  cvt(Whh2, Whh2b, 768 * 256);
  cvt(Wo, Wob, OUTD * 256);

  prenet_kernel<<<(N_BATCH * T_STEPS) / 16, 256, 0, stream>>>(inputs, W1b, b1,
                                                              W2b, b2, pre);
  pm_kernel<<<(N_BATCH * TX) / 16, 256, 0, stream>>>(memory, Wmb, pmb);

  float* outs   = (float*)d_out;
  float* aligns = outs + (size_t)N_BATCH * T_STEPS * OUTD;
  size_t lds = ((size_t)(2 * 16 * 768 + 7 * 16 * 256 + 256) * sizeof(float)) +
               ((size_t)(16 * 512 + 16 * 256) * sizeof(bf16));
  decoder_kernel<<<N_BATCH / 16, 512, lds, stream>>>(
      pre, pmb, memory, v,
      Wihab, bih_a, Whhab, bhh_a, Wqb, Wpb, bp,
      Wih1b, bih1, Whh1b, bhh1, Wih2b, bih2, Whh2b, bhh2, Wob, bo,
      outs, aligns);
}
